// MultiHeadAttention_65197603553384
// MI455X (gfx1250) — matrix operand order repaired and checked by reading, not compiled, not run on hardware
//
#include <hip/hip_runtime.h>

// ---------------- types ----------------
typedef __attribute__((ext_vector_type(16))) __bf16 v16bf;
typedef __attribute__((ext_vector_type(8)))  __bf16 v8bf;
typedef __attribute__((ext_vector_type(8)))  float  v8f;

static __device__ __forceinline__ v8f wmma_bf16(v16bf a, v16bf b, v8f c) {
  return __builtin_amdgcn_wmma_f32_16x16x32_bf16(false, a, false, b, (short)0, c, false, false);
}

static __device__ __forceinline__ v16bf join16(v8bf lo, v8bf hi) {
  union { v16bf v; v8bf h[2]; } u;
  u.h[0] = lo; u.h[1] = hi;
  return u.v;
}

// problem constants
#define DM    1024
#define NH    16
#define HD    64
#define BB    2
#define SS    2048
#define MROWS (BB * SS)      // 4096
#define LN_THETA 9.2103403719761836f  // ln(10000)
#define LDSW  40             // padded LDS row stride (halves): 80B = 20 banks -> conflict-free frag reads

// CDNA5 async global->LDS copy, if the toolchain exposes it.
// Probe-derived signature: (v4i addrspace(1)*, v4i addrspace(3)*, imm offset, imm cpol)
#if __has_builtin(__builtin_amdgcn_global_load_async_to_lds_b128)
#define HAVE_ASYNC_LDS 1
typedef int v4i_vs __attribute__((vector_size(16)));
typedef __attribute__((address_space(1))) v4i_vs* async_gp;
typedef __attribute__((address_space(3))) v4i_vs* async_lp;
#endif

static __device__ __forceinline__ void async_fence_lds() {
#if defined(HAVE_ASYNC_LDS)
#if __has_builtin(__builtin_amdgcn_s_wait_asynccnt)
  __builtin_amdgcn_s_wait_asynccnt(0);
#else
  asm volatile("s_wait_asynccnt 0" ::: "memory");
#endif
#endif
  // fallback path uses ds_store; __syncthreads() (dscnt fence) provides visibility
}

// ---------------- fp32 -> bf16 convert ----------------
__global__ void cvt_f32_bf16(const float* __restrict__ src, __bf16* __restrict__ dst, int n) {
  for (int i = blockIdx.x * blockDim.x + threadIdx.x; i < n; i += gridDim.x * blockDim.x)
    dst[i] = (__bf16)src[i];
}

// cooperative stage of W tile [64 n x 32 k] bf16 into padded LDS (4KB payload)
static __device__ __forceinline__ void stage_tile(const __bf16* __restrict__ W,
                                                  __bf16* tile, int n0, int k0, int tid) {
  const int row = tid >> 2;   // 0..63
  const int ch  = tid & 3;    // 0..3 (16B chunks of the 64B row)
  const __bf16* g = W + (size_t)(n0 + row) * DM + k0 + ch * 8;
  __bf16* l = tile + row * LDSW + ch * 8;
#if defined(HAVE_ASYNC_LDS)
  __builtin_amdgcn_global_load_async_to_lds_b128((async_gp)g, (async_lp)l, 0, 0);
#else
  v8bf v = *(const v8bf*)g;
  *(v8bf*)l = v;
#endif
}

// ---------------- WMMA GEMM: C[M=4096,N=1024] = A(bf16,row-major) @ W^T(bf16,[N,K] row-major)
// block = 8 waves, tile 256(M) x 64(N); W k-slab double-buffered in LDS, shared by all waves.
// mode 0: RoPE epilogue, store bf16 to dst[B,H,S,64]
// mode 1: store bf16 transposed to dst[B,H,64,S]
// mode 2: store fp32 row-major dst[M,N]
__global__ void gemm_wmma(const __bf16* __restrict__ A, const __bf16* __restrict__ W,
                          const int* __restrict__ tokpos, void* __restrict__ dst, int mode) {
  const int tid    = threadIdx.x;
  const int lane   = tid & 31;
  const int lane15 = lane & 15;
  const int h2     = lane >> 4;
  const int wave   = tid >> 5;

  const int m0 = blockIdx.x * 256 + wave * 32;  // 32 rows per wave
  const int n0 = blockIdx.y * 64;               // 64 cols per block (4 tiles)

  __shared__ __align__(16) __bf16 wtile[2][64 * LDSW];

  v8f acc[2][4] = {};
  const __bf16* arow0 = A + (size_t)(m0 + lane15) * DM;
  const __bf16* arow1 = arow0 + (size_t)16 * DM;

  stage_tile(W, wtile[0], n0, 0, tid);
  async_fence_lds();
  __syncthreads();

  for (int k0 = 0; k0 < DM; k0 += 32) {
    const int buf = (k0 >> 5) & 1;
    if (k0 + 32 < DM) stage_tile(W, wtile[buf ^ 1], n0, k0 + 32, tid);

    // A fragments (global, unique per wave): halves 0-7 -> k0+h2*8+i, 8-15 -> k0+16+h2*8+i
    v16bf a0 = join16(*(const v8bf*)(arow0 + k0 + h2 * 8),
                      *(const v8bf*)(arow0 + k0 + 16 + h2 * 8));
    v16bf a1 = join16(*(const v8bf*)(arow1 + k0 + h2 * 8),
                      *(const v8bf*)(arow1 + k0 + 16 + h2 * 8));

    // B fragments from LDS (shared by all 8 waves)
    const __bf16* bt = wtile[buf];
#pragma unroll
    for (int t = 0; t < 4; ++t) {
      const __bf16* bp = bt + (t * 16 + lane15) * LDSW + h2 * 8;
      v16bf b = join16(*(const v8bf*)bp, *(const v8bf*)(bp + 16));
      acc[0][t] = wmma_bf16(a0, b, acc[0][t]);
      acc[1][t] = wmma_bf16(a1, b, acc[1][t]);
    }

    async_fence_lds();
    __syncthreads();
  }

  const int head = n0 >> 6;  // each 64-wide strip is exactly one head

#pragma unroll
  for (int mt = 0; mt < 2; ++mt) {
    const int mb = m0 + mt * 16;
    if (mode == 0) {
      // RoPE + store bf16 [B,H,S,HD]
      __bf16* out = (__bf16*)dst;
#pragma unroll
      for (int t = 0; t < 4; ++t) {
        const int dc = t * 16 + lane15;       // head-dim column 0..63
        const int i2 = dc >> 1;               // rotary pair index
        const float freq = __expf(-(float)i2 * (LN_THETA / 32.0f)); // theta^(-2*i2/64)
#pragma unroll
        for (int r = 0; r < 8; ++r) {
          const int m = mb + r + 8 * h2;
          const int b = m >> 11;
          const int s = m & (SS - 1);
          const float pos = (float)tokpos[s];
          float sn, cs;
          __sincosf(pos * freq, &sn, &cs);
          const float val = acc[mt][t][r];
          const float pv  = __shfl_xor(val, 1, 32);  // even/odd rotary partner
          const float res = ((lane15 & 1) == 0) ? (val * cs - pv * sn)
                                                : (pv * sn + val * cs);
          out[((size_t)(b * NH + head) * SS + s) * HD + dc] = (__bf16)res;
        }
      }
    } else if (mode == 1) {
      // V stored transposed: [B,H,HD,S]
      __bf16* out = (__bf16*)dst;
#pragma unroll
      for (int t = 0; t < 4; ++t) {
        const int dc = t * 16 + lane15;
#pragma unroll
        for (int r = 0; r < 8; ++r) {
          const int m = mb + r + 8 * h2;
          const int b = m >> 11;
          const int s = m & (SS - 1);
          out[((size_t)(b * NH + head) * HD + dc) * SS + s] = (__bf16)acc[mt][t][r];
        }
      }
    } else {
      // plain fp32 row-major [M,N]
      float* out = (float*)dst;
#pragma unroll
      for (int t = 0; t < 4; ++t) {
        const int n = n0 + t * 16 + lane15;
#pragma unroll
        for (int r = 0; r < 8; ++r) {
          const int m = mb + r + 8 * h2;
          out[(size_t)m * DM + n] = acc[mt][t][r];
        }
      }
    }
  }
}

// ---------------- Flash attention (causal), one wave = 16 query rows ----------------
__global__ void attn_wmma(const __bf16* __restrict__ Qb, const __bf16* __restrict__ Kb,
                          const __bf16* __restrict__ VTb, __bf16* __restrict__ ctx) {
  const int lane   = threadIdx.x & 31;
  const int lane15 = lane & 15;
  const int h2     = lane >> 4;
  const int wave   = threadIdx.x >> 5;

  const int nqc = SS / 64;                 // 32 query chunks (64 rows) per (b,h)
  const int bh  = blockIdx.x / nqc;
  const int qc  = blockIdx.x % nqc;
  const int q0  = qc * 64 + wave * 16;
  const int b   = bh / NH;
  const int head = bh % NH;

  const __bf16* Qh = Qb  + (size_t)bh * SS * HD;
  const __bf16* Kh = Kb  + (size_t)bh * SS * HD;
  const __bf16* Vh = VTb + (size_t)bh * HD * SS;

  __shared__ __align__(16) __bf16 plds[4][16 * 32];
  __bf16* myp = plds[wave];

  // Q fragments, hoisted (d 0..31 and d 32..63)
  const __bf16* qrow = Qh + (size_t)(q0 + lane15) * HD;
  v16bf qa0 = *(const v16bf*)(qrow + h2 * 16);
  v16bf qa1 = *(const v16bf*)(qrow + 32 + h2 * 16);

  float mrow[8], lrow[8];
  v8f oacc[4] = {};
#pragma unroll
  for (int r = 0; r < 8; ++r) { mrow[r] = -1e30f; lrow[r] = 0.0f; }

  for (int k0 = 0; k0 < q0 + 16; k0 += 32) {
    // K fragments: key = base + lane15, d runs contiguous
    const __bf16* krow0 = Kh + (size_t)(k0 + lane15) * HD + h2 * 16;
    const __bf16* krow1 = Kh + (size_t)(k0 + 16 + lane15) * HD + h2 * 16;
    v16bf kb00 = *(const v16bf*)(krow0);
    v16bf kb01 = *(const v16bf*)(krow0 + 32);
    v16bf kb10 = *(const v16bf*)(krow1);
    v16bf kb11 = *(const v16bf*)(krow1 + 32);

    v8f s0 = {}, s1 = {};
    s0 = wmma_bf16(qa0, kb00, s0);
    s0 = wmma_bf16(qa1, kb01, s0);
    s1 = wmma_bf16(qa0, kb10, s1);
    s1 = wmma_bf16(qa1, kb11, s1);

    const bool needmask = (k0 + 31 > q0);

#pragma unroll
    for (int r = 0; r < 8; ++r) {
      float x0 = s0[r] * 0.125f;   // 1/sqrt(64)
      float x1 = s1[r] * 0.125f;
      if (needmask) {
        const int qr = q0 + r + 8 * h2;
        if (k0 + lane15 > qr)      x0 = -1e30f;
        if (k0 + 16 + lane15 > qr) x1 = -1e30f;
      }
      // row max over 32 keys (16-lane butterfly inside each half)
      float rm = fmaxf(x0, x1);
      rm = fmaxf(rm, __shfl_xor(rm, 1, 32));
      rm = fmaxf(rm, __shfl_xor(rm, 2, 32));
      rm = fmaxf(rm, __shfl_xor(rm, 4, 32));
      rm = fmaxf(rm, __shfl_xor(rm, 8, 32));
      const float mnew  = fmaxf(mrow[r], rm);
      const float alpha = __expf(mrow[r] - mnew);
      mrow[r] = mnew;
      const float e0 = __expf(x0 - mnew);
      const float e1 = __expf(x1 - mnew);
      float rs = e0 + e1;
      rs += __shfl_xor(rs, 1, 32);
      rs += __shfl_xor(rs, 2, 32);
      rs += __shfl_xor(rs, 4, 32);
      rs += __shfl_xor(rs, 8, 32);
      lrow[r] = alpha * lrow[r] + rs;
      oacc[0][r] *= alpha; oacc[1][r] *= alpha;
      oacc[2][r] *= alpha; oacc[3][r] *= alpha;
      // spill P (C layout) to per-wave LDS tile (row-major 16x32 bf16)
      const int prow = r + 8 * h2;
      myp[prow * 32 + lane15]      = (__bf16)e0;
      myp[prow * 32 + 16 + lane15] = (__bf16)e1;
    }

    // same-wave LDS RAW: wait ds ops, then reload in A-fragment layout
    asm volatile("s_wait_dscnt 0" ::: "memory");
    v8bf plo = *(const v8bf*)(myp + lane15 * 32 + h2 * 16);
    v8bf phi = *(const v8bf*)(myp + lane15 * 32 + 8 + h2 * 16);
    v16bf pa = join16(plo, phi);

    // P(16x32) @ V(32x64): V^T rows are contiguous in key
#pragma unroll
    for (int t = 0; t < 4; ++t) {
      v16bf vb = *(const v16bf*)(Vh + (size_t)(t * 16 + lane15) * SS + k0 + h2 * 16);
      oacc[t] = wmma_bf16(pa, vb, oacc[t]);
    }
  }

  // normalize and store context bf16 as [B,S,D] (head-interleaved row-major)
#pragma unroll
  for (int r = 0; r < 8; ++r) {
    const float inv = 1.0f / lrow[r];
    const int qr = q0 + r + 8 * h2;
#pragma unroll
    for (int t = 0; t < 4; ++t) {
      const int dc = t * 16 + lane15;
      ctx[((size_t)(b * SS + qr)) * DM + head * HD + dc] = (__bf16)(oacc[t][r] * inv);
    }
  }
}

// ---------------- host side ----------------
extern "C" void kernel_launch(void* const* d_in, const int* in_sizes, int n_in,
                              void* d_out, int out_size, void* d_ws, size_t ws_size,
                              hipStream_t stream) {
  const float* x      = (const float*)d_in[0];
  const int*   tokpos = (const int*)d_in[1];
  const float* Wq     = (const float*)d_in[2];
  const float* Wk     = (const float*)d_in[3];
  const float* Wv     = (const float*)d_in[4];
  const float* Wo     = (const float*)d_in[5];

  char* ws = (char*)d_ws;
  __bf16* xb   = (__bf16*)(ws);                        // 8 MiB
  __bf16* wqb  = (__bf16*)(ws + ((size_t)8  << 20));   // 2 MiB
  __bf16* wkb  = (__bf16*)(ws + ((size_t)10 << 20));
  __bf16* wvb  = (__bf16*)(ws + ((size_t)12 << 20));
  __bf16* wob  = (__bf16*)(ws + ((size_t)14 << 20));
  __bf16* qbuf = (__bf16*)(ws + ((size_t)16 << 20));   // 8 MiB [B,H,S,64]
  __bf16* kbuf = (__bf16*)(ws + ((size_t)24 << 20));   // 8 MiB
  __bf16* vtbf = (__bf16*)(ws + ((size_t)32 << 20));   // 8 MiB [B,H,64,S]
  __bf16* ctx  = (__bf16*)(ws + ((size_t)40 << 20));   // 8 MiB [B,S,D]

  // fp32 -> bf16 conversions
  cvt_f32_bf16<<<2048, 256, 0, stream>>>(x,  xb,  MROWS * DM);
  cvt_f32_bf16<<<1024, 256, 0, stream>>>(Wq, wqb, DM * DM);
  cvt_f32_bf16<<<1024, 256, 0, stream>>>(Wk, wkb, DM * DM);
  cvt_f32_bf16<<<1024, 256, 0, stream>>>(Wv, wvb, DM * DM);
  cvt_f32_bf16<<<1024, 256, 0, stream>>>(Wo, wob, DM * DM);

  // QKV projections (RoPE fused for Q/K, transposed store for V)
  dim3 ggrid(MROWS / 256, DM / 64);
  gemm_wmma<<<ggrid, 256, 0, stream>>>(xb, wqb, tokpos, qbuf, 0);
  gemm_wmma<<<ggrid, 256, 0, stream>>>(xb, wkb, tokpos, kbuf, 0);
  gemm_wmma<<<ggrid, 256, 0, stream>>>(xb, wvb, tokpos, vtbf, 1);

  // causal flash attention: B*H*(S/64) blocks, 4 waves (64 query rows) each
  attn_wmma<<<BB * NH * (SS / 64), 128, 0, stream>>>(qbuf, kbuf, vtbf, ctx);

  // output projection -> fp32 d_out
  gemm_wmma<<<ggrid, 256, 0, stream>>>(ctx, wob, tokpos, (float*)d_out, 2);
}
